// GNN_4647154614414
// MI455X (gfx1250) — matrix operand order repaired and checked by reading, not compiled, not run on hardware
//
#include <hip/hip_runtime.h>
#include <hip/hip_bf16.h>
#include <stdint.h>

static constexpr int KD  = 128;  // node feature dim
static constexpr int KEF = 16;   // edge feature dim
static constexpr int KL  = 3;    // layers

typedef __attribute__((ext_vector_type(16))) __bf16 v16bf;
typedef __attribute__((ext_vector_type(8)))  float  v8f;

#if defined(__has_builtin)
#if __has_builtin(__builtin_amdgcn_sched_barrier)
#define SCHED_FENCE() __builtin_amdgcn_sched_barrier(0)
#endif
#endif
#ifndef SCHED_FENCE
#define SCHED_FENCE()
#endif

union FragB16 {
    v16bf v;
    uint4 q[2];
    unsigned short h[16];
};

__device__ __forceinline__ unsigned short f2bf(float f) {
    unsigned int u = __float_as_uint(f);
    u += 0x7FFFu + ((u >> 16) & 1u);   // round-to-nearest-even
    return (unsigned short)(u >> 16);
}

// ---------------------------------------------------------------------------
// Weight fragment swizzle: f32 [L, Ksrc, 128] -> bf16 fragments
// dst layout: [L][NC][8 ntiles][32 lanes][16 halves]
//   B element for lane t, half i of chunk c, ntile j:
//     k = c*32 + 8*(t>>4) + (i&7) + 16*(i>>3)   (32x16 B layout: lanes 16-31 hold K+8, halves 8-15 hold K+16)
//     n = j*16 + (t&15)
//   k >= Ksrc  -> zero padding (handles K=272 -> 288)
// ---------------------------------------------------------------------------
__global__ void gnn_wconv_kernel(const float* __restrict__ src,
                                 unsigned short* __restrict__ dst,
                                 int NC, int Ksrc, int total) {
    int i = blockIdx.x * blockDim.x + threadIdx.x;
    int stride = gridDim.x * blockDim.x;
    for (; i < total; i += stride) {
        int ih   = i & 15;
        int t    = (i >> 4) & 31;
        int j    = (i >> 9) & 7;
        int rest = i >> 12;
        int c    = rest % NC;
        int l    = rest / NC;
        int k = c * 32 + 8 * (t >> 4) + (ih & 7) + 16 * (ih >> 3);
        int n = j * 16 + (t & 15);
        float v = (k < Ksrc) ? src[((size_t)l * Ksrc + k) * KD + n] : 0.0f;
        dst[i] = f2bf(v);
    }
}

// ---------------------------------------------------------------------------
// Per-layer prep (vectorized): bf16 shadow of h + zero reduction buffer
// ---------------------------------------------------------------------------
__global__ void gnn_prep_kernel(const float4* __restrict__ h,
                                ushort4* __restrict__ hbf,
                                float4* __restrict__ red, size_t total4) {
    size_t i = (size_t)blockIdx.x * blockDim.x + threadIdx.x;
    size_t stride = (size_t)gridDim.x * blockDim.x;
    for (; i < total4; i += stride) {
        float4 v = h[i];
        ushort4 o;
        o.x = f2bf(v.x); o.y = f2bf(v.y); o.z = f2bf(v.z); o.w = f2bf(v.w);
        hbf[i] = o;
        red[i] = make_float4(0.0f, 0.0f, 0.0f, 0.0f);
    }
}

// Load 8 B fragments (one K-chunk, all 8 N-tiles), fenced so the scheduler
// cannot sink the loads into the WMMA burst.
__device__ __forceinline__ void load_b8(const unsigned short* __restrict__ wb,
                                        FragB16 (&Bf)[8]) {
    #pragma unroll
    for (int j = 0; j < 8; ++j) {
        Bf[j].q[0] = *(const uint4*)(wb + (size_t)j * 512);
        Bf[j].q[1] = *(const uint4*)(wb + (size_t)j * 512 + 8);
    }
    SCHED_FENCE();
}

// ---------------------------------------------------------------------------
// Edge kernel: one wave = 16 edges. msg = leaky(e_in @ W1) @ W2, atomic scatter.
//   e_in K = 288 padded (src h: chunks 0-3, dst h: 4-7, ef+pad: 8)
//   A layout (16-bit 16x32): lane t = row t%16; halves 0..7 -> K0..K0+7,
//   halves 8..15 -> K0+16..K0+23 with K0 = 32c + 8*(t>>4)  => two b128 loads.
// ---------------------------------------------------------------------------
__global__ __launch_bounds__(128)
__attribute__((amdgpu_waves_per_eu(1, 8)))
void gnn_edge_kernel(
    const unsigned short* __restrict__ hbf,
    const float* __restrict__ ef,
    const int* __restrict__ src,
    const int* __restrict__ dst,
    const unsigned short* __restrict__ W1f,   // [9][8][32][16]
    const unsigned short* __restrict__ W2f,   // [4][8][32][16]
    float* __restrict__ red,
    int E, int ntile)
{
    __shared__ unsigned short lH[4][16 * 136];   // padded: bank-conflict-free
    __shared__ int lD[4][16];

    const int wave = threadIdx.x >> 5;
    const int t    = threadIdx.x & 31;
    const int tile = blockIdx.x * 4 + wave;
    const bool active = tile < ntile;            // wave-uniform (EXEC stays full)
    const int m    = t & 15;
    const int half = t >> 4;
    const int r0   = half * 8;
    unsigned short* Lh = lH[wave];

    if (active) {
        const int e0 = tile * 16;
        int e = e0 + m;
        int s_i, d_i, erow;
        if (e < E) { s_i = src[e]; d_i = dst[e]; erow = e; }
        else       { int e2 = e - E; s_i = dst[e2]; d_i = src[e2]; erow = e2; }
        if (t < 16) lD[wave][m] = d_i;

        const unsigned short* srow = hbf + (size_t)s_i * KD;
        const unsigned short* drow = hbf + (size_t)d_i * KD;

        v8f acc[8] = {};
        #pragma unroll
        for (int c = 0; c < 9; ++c) {
            FragB16 A;
            if (c < 8) {
                const unsigned short* base = (c < 4) ? srow : drow;
                const int k0 = (c & 3) * 32 + half * 8;
                A.q[0] = *(const uint4*)(base + k0);
                A.q[1] = *(const uint4*)(base + k0 + 16);
            } else {
                const float* er = ef + (size_t)erow * KEF + half * 8;
                #pragma unroll
                for (int i = 0; i < 8; ++i) A.h[i] = f2bf(er[i]);
                A.q[1] = make_uint4(0u, 0u, 0u, 0u);   // K 272..287 zero pad
            }
            FragB16 Bf[8];
            load_b8(W1f + ((size_t)(c * 8) * 32 + t) * 16, Bf);
            #pragma unroll
            for (int j = 0; j < 8; ++j) {
                acc[j] = __builtin_amdgcn_wmma_f32_16x16x32_bf16(
                    false, A.v, false, Bf[j].v, (short)0, acc[j], false, false);
            }
        }
        // LeakyReLU(0.01) + transpose through LDS (C layout -> A layout)
        #pragma unroll
        for (int j = 0; j < 8; ++j) {
            #pragma unroll
            for (int r = 0; r < 8; ++r) {
                float v = acc[j][r];
                v = v > 0.0f ? v : 0.01f * v;
                Lh[(r + r0) * 136 + (m + 16 * j)] = f2bf(v);
            }
        }
    }
    __syncthreads();
    if (active) {
        v8f acc2[8] = {};
        #pragma unroll
        for (int c = 0; c < 4; ++c) {
            FragB16 A;
            const int k0 = c * 32 + half * 8;
            A.q[0] = *(const uint4*)(Lh + m * 136 + k0);
            A.q[1] = *(const uint4*)(Lh + m * 136 + k0 + 16);
            FragB16 Bf[8];
            load_b8(W2f + ((size_t)(c * 8) * 32 + t) * 16, Bf);
            #pragma unroll
            for (int j = 0; j < 8; ++j) {
                acc2[j] = __builtin_amdgcn_wmma_f32_16x16x32_bf16(
                    false, A.v, false, Bf[j].v, (short)0, acc2[j], false, false);
            }
        }
        // segment-sum via f32 global atomics
        #pragma unroll
        for (int j = 0; j < 8; ++j) {
            const int n = m + 16 * j;
            #pragma unroll
            for (int r = 0; r < 8; ++r) {
                const int di = lD[wave][r + r0];
                atomicAdd(red + (size_t)di * KD + n, acc2[j][r]);
            }
        }
    }
}

// ---------------------------------------------------------------------------
// Node kernel: one wave = 16 nodes. out = leaky([h,red] @ W1) @ W2 (+ h)
// ---------------------------------------------------------------------------
__global__ __launch_bounds__(128)
__attribute__((amdgpu_waves_per_eu(1, 8)))
void gnn_node_kernel(
    const unsigned short* __restrict__ hbf,
    const float* __restrict__ red,
    const float* __restrict__ hres,
    const unsigned short* __restrict__ W1f,   // [8][8][32][16]
    const unsigned short* __restrict__ W2f,   // [4][8][32][16]
    float* __restrict__ outp,
    int ntile, int do_res)
{
    __shared__ unsigned short lH[4][16 * 136];

    const int wave = threadIdx.x >> 5;
    const int t    = threadIdx.x & 31;
    const int tile = blockIdx.x * 4 + wave;
    const bool active = tile < ntile;
    const int m    = t & 15;
    const int half = t >> 4;
    const int r0   = half * 8;
    unsigned short* Lh = lH[wave];
    const int n0 = tile * 16;

    if (active) {
        const int node = n0 + m;
        const unsigned short* hrow = hbf + (size_t)node * KD;
        const float* rrow = red + (size_t)node * KD;

        v8f acc[8] = {};
        #pragma unroll
        for (int c = 0; c < 8; ++c) {
            FragB16 A;
            if (c < 4) {
                const int k0 = c * 32 + half * 8;
                A.q[0] = *(const uint4*)(hrow + k0);
                A.q[1] = *(const uint4*)(hrow + k0 + 16);
            } else {
                const int k0 = (c - 4) * 32 + half * 8;
                #pragma unroll
                for (int i = 0; i < 8; ++i) A.h[i]     = f2bf(rrow[k0 + i]);
                #pragma unroll
                for (int i = 0; i < 8; ++i) A.h[8 + i] = f2bf(rrow[k0 + 16 + i]);
            }
            FragB16 Bf[8];
            load_b8(W1f + ((size_t)(c * 8) * 32 + t) * 16, Bf);
            #pragma unroll
            for (int j = 0; j < 8; ++j) {
                acc[j] = __builtin_amdgcn_wmma_f32_16x16x32_bf16(
                    false, A.v, false, Bf[j].v, (short)0, acc[j], false, false);
            }
        }
        #pragma unroll
        for (int j = 0; j < 8; ++j) {
            #pragma unroll
            for (int r = 0; r < 8; ++r) {
                float v = acc[j][r];
                v = v > 0.0f ? v : 0.01f * v;
                Lh[(r + r0) * 136 + (m + 16 * j)] = f2bf(v);
            }
        }
    }
    __syncthreads();
    if (active) {
        v8f acc2[8] = {};
        #pragma unroll
        for (int c = 0; c < 4; ++c) {
            FragB16 A;
            const int k0 = c * 32 + half * 8;
            A.q[0] = *(const uint4*)(Lh + m * 136 + k0);
            A.q[1] = *(const uint4*)(Lh + m * 136 + k0 + 16);
            FragB16 Bf[8];
            load_b8(W2f + ((size_t)(c * 8) * 32 + t) * 16, Bf);
            #pragma unroll
            for (int j = 0; j < 8; ++j) {
                acc2[j] = __builtin_amdgcn_wmma_f32_16x16x32_bf16(
                    false, A.v, false, Bf[j].v, (short)0, acc2[j], false, false);
            }
        }
        #pragma unroll
        for (int j = 0; j < 8; ++j) {
            const int n = m + 16 * j;
            #pragma unroll
            for (int r = 0; r < 8; ++r) {
                const int row = n0 + r + r0;
                float v = acc2[j][r];
                if (do_res) v += hres[(size_t)row * KD + n];
                outp[(size_t)row * KD + n] = v;
            }
        }
    }
}

// ---------------------------------------------------------------------------
extern "C" void kernel_launch(void* const* d_in, const int* in_sizes, int n_in,
                              void* d_out, int out_size, void* d_ws, size_t ws_size,
                              hipStream_t stream) {
    const float* nf  = (const float*)d_in[0];
    const float* ef  = (const float*)d_in[1];
    const int*   src = (const int*)d_in[2];
    const int*   dst = (const int*)d_in[3];
    const float* We1 = (const float*)d_in[4];
    const float* We2 = (const float*)d_in[5];
    const float* Wn1 = (const float*)d_in[6];
    const float* Wn2 = (const float*)d_in[7];
    float* out = (float*)d_out;

    const int N    = in_sizes[0] / KD;
    const int E    = in_sizes[2];
    const int twoE = 2 * E;
    (void)n_in; (void)out_size; (void)ws_size;

    char* ws = (char*)d_ws;
    size_t off = 0;
    auto walloc = [&](size_t bytes) -> void* {
        void* p = ws + off;
        off += bytes;
        off = (off + 255) & ~(size_t)255;
        return p;
    };
    unsigned short* hbf  = (unsigned short*)walloc((size_t)N * KD * 2);
    float*          red  = (float*)walloc((size_t)N * KD * 4);
    float*          bufA = (float*)walloc((size_t)N * KD * 4);
    float*          bufB = (float*)walloc((size_t)N * KD * 4);
    unsigned short* We1f = (unsigned short*)walloc((size_t)KL * 9 * 8 * 512 * 2);
    unsigned short* We2f = (unsigned short*)walloc((size_t)KL * 4 * 8 * 512 * 2);
    unsigned short* Wn1f = (unsigned short*)walloc((size_t)KL * 8 * 8 * 512 * 2);
    unsigned short* Wn2f = (unsigned short*)walloc((size_t)KL * 4 * 8 * 512 * 2);

    // Pre-swizzle weights to WMMA B-fragment order (bf16)
    int tot;
    tot = KL * 9 * 8 * 512;
    gnn_wconv_kernel<<<(tot + 255) / 256, 256, 0, stream>>>(We1, We1f, 9, 2 * KD + KEF, tot);
    tot = KL * 4 * 8 * 512;
    gnn_wconv_kernel<<<(tot + 255) / 256, 256, 0, stream>>>(We2, We2f, 4, KD, tot);
    tot = KL * 8 * 8 * 512;
    gnn_wconv_kernel<<<(tot + 255) / 256, 256, 0, stream>>>(Wn1, Wn1f, 8, 2 * KD, tot);
    tot = KL * 4 * 8 * 512;
    gnn_wconv_kernel<<<(tot + 255) / 256, 256, 0, stream>>>(Wn2, Wn2f, 4, KD, tot);

    const int etile = (twoE + 15) / 16;
    const int ntile = (N + 15) / 16;
    const size_t nd4 = (size_t)N * KD / 4;

    const float* hcur = nf;
    for (int l = 0; l < KL; ++l) {
        gnn_prep_kernel<<<2048, 256, 0, stream>>>(
            (const float4*)hcur, (ushort4*)hbf, (float4*)red, nd4);
        gnn_edge_kernel<<<(etile + 3) / 4, 128, 0, stream>>>(
            hbf, ef, src, dst,
            We1f + (size_t)l * 9 * 8 * 512,
            We2f + (size_t)l * 4 * 8 * 512,
            red, E, etile);
        float* hnext = (l == KL - 1) ? out : ((l == 0) ? bufA : bufB);
        gnn_node_kernel<<<(ntile + 3) / 4, 128, 0, stream>>>(
            hbf, red, hcur,
            Wn1f + (size_t)l * 8 * 8 * 512,
            Wn2f + (size_t)l * 4 * 8 * 512,
            hnext, ntile, (l < KL - 1) ? 1 : 0);
        hcur = hnext;
    }
}
